// SwinUNet_84628035600613
// MI455X (gfx1250) — matrix operand order repaired and checked by reading, not compiled, not run on hardware
//
#include <hip/hip_runtime.h>

#define DEVINL __device__ __forceinline__

typedef __bf16 bf16_t;
typedef bf16_t v16bf __attribute__((ext_vector_type(16)));
typedef float  v8f   __attribute__((ext_vector_type(8)));

#define HAS_TDM __has_builtin(__builtin_amdgcn_tensor_load_to_lds)

#if HAS_TDM
typedef unsigned int v4u __attribute__((ext_vector_type(4)));
typedef int          v8i __attribute__((ext_vector_type(8)));
typedef int          v4i __attribute__((ext_vector_type(4)));
#endif

union Frag {
    v16bf v;
    int4  i2[2];
};

DEVINL unsigned short f2bf(float f) {
    unsigned u = __float_as_uint(f);
    u += 0x7fffu + ((u >> 16) & 1u);   // round-to-nearest-even
    return (unsigned short)(u >> 16);
}

DEVINL v8f vzero8() {
    v8f z;
#pragma unroll
    for (int i = 0; i < 8; ++i) z[i] = 0.0f;
    return z;
}

DEVINL v8f wmma_bf16(v16bf a, v16bf b, v8f c) {
    // D = A(16x32 bf16) * B(32x16 bf16) + C(16x16 f32)
    return __builtin_amdgcn_wmma_f32_16x16x32_bf16(false, a, false, b, (short)0, c, false, false);
}

#if HAS_TDM
DEVINL unsigned lds_off(const void* p) {
    // LDS aperture: flat address low 32 bits = LDS byte address
    return (unsigned)(unsigned long long)p;
}

// Issue a TDM 2D tile load: tile (tileD1 rows x tileD0 elems) of 2-byte data,
// row stride = stride0 elems, into LDS at ldsAddr with +16B padding every 64B
// (reproduces the 40-halfword padded LDS row stride used by the WMMA frag reads).
DEVINL void tdm_load_2d(unsigned ldsAddr, const void* gptr,
                        unsigned tensorD0, unsigned tensorD1,
                        unsigned tileD0, unsigned tileD1, unsigned stride0) {
    unsigned long long ga = (unsigned long long)gptr;
    v4u g0 = { 1u,                                   // count=1, user mode
               ldsAddr,                              // lds_addr
               (unsigned)ga,                         // global_addr[31:0]
               (unsigned)((ga >> 32) & 0x1ffffffu) | (2u << 30) };  // [56:32] | type=2
    v8i g1;
    // data_size=2B(1<<16) | pad_enable(1<<20) | pad_interval=16dw(3<<22) | pad_amount=4dw(3<<25)
    g1[0] = (int)((1u << 16) | (1u << 20) | (3u << 22) | (3u << 25));
    g1[1] = (int)((tensorD0 & 0xffffu) << 16);                                  // dim0[15:0] @ bits63:48
    g1[2] = (int)(((tensorD0 >> 16) & 0xffffu) | ((tensorD1 & 0xffffu) << 16)); // dim0[31:16], dim1[15:0]
    g1[3] = (int)(((tensorD1 >> 16) & 0xffffu) | (tileD0 << 16));               // dim1[31:16], tile_dim0
    g1[4] = (int)tileD1;                                                        // tile_dim1 (tile_dim2=0)
    g1[5] = (int)stride0;                                                       // dim0_stride[31:0]
    g1[6] = 0;
    g1[7] = 0;
    v4i z4 = {0, 0, 0, 0};
#if __clang_major__ >= 23
    v8i z8 = {0, 0, 0, 0, 0, 0, 0, 0};
    __builtin_amdgcn_tensor_load_to_lds(g0, g1, z4, z4, z8, 0);
#else
    __builtin_amdgcn_tensor_load_to_lds(g0, g1, z4, z4, 0);
#endif
}
#endif  // HAS_TDM

// ---------------------------------------------------------------------------
// K0: convert weight [K x N] f32 row-major -> [N x K] bf16 (transposed)
// ---------------------------------------------------------------------------
__global__ void wconvT_kernel(const float* __restrict__ w,
                              unsigned short* __restrict__ o, int K, int N) {
    int i = blockIdx.x * 256 + threadIdx.x;
    if (i >= N * K) return;
    int n = i / K, k = i - n * K;
    o[i] = f2bf(w[(size_t)k * N + n]);
}

// ---------------------------------------------------------------------------
// K1: mod[b][j] = silu(emb[b]) @ ada_w + ada_b     (8 x 3072, K=512)
// ---------------------------------------------------------------------------
__global__ void __launch_bounds__(256) mod_kernel(const float* __restrict__ emb,
                                                  const float* __restrict__ ada_w,
                                                  const float* __restrict__ ada_b,
                                                  float* __restrict__ mod) {
    __shared__ float se[512];
    int b = blockIdx.x / 12;
    int j = (blockIdx.x % 12) * 256 + threadIdx.x;
    float e0 = emb[b * 512 + threadIdx.x];
    float e1 = emb[b * 512 + 256 + threadIdx.x];
    se[threadIdx.x]       = e0 / (1.f + __expf(-e0));
    se[threadIdx.x + 256] = e1 / (1.f + __expf(-e1));
    __syncthreads();
    float acc = ada_b[j];
    for (int c = 0; c < 512; ++c)
        acc += se[c] * ada_w[(size_t)c * 3072 + j];
    mod[b * 3072 + j] = acc;
}

// ---------------------------------------------------------------------------
// K2/K6: LayerNorm (no affine, eps 1e-6) + (1+sc)*h + sh -> bf16
// mode 0: dest row = window-major (roll -4,-4 + 8x8 partition), src = x
// mode 1: dest row = token-major, src = x1 (d_out)
// ---------------------------------------------------------------------------
__global__ void __launch_bounds__(256) ln_mod_kernel(const float* __restrict__ src,
                                                     const float* __restrict__ mod,
                                                     unsigned short* __restrict__ dst,
                                                     int shOff, int scOff, int mode) {
    int wave = threadIdx.x >> 5, lane = threadIdx.x & 31;
    int r = blockIdx.x * 8 + wave;
    int b, srcTok;
    if (mode == 0) {
        int win = r >> 6, n = r & 63;
        b = win >> 6;
        int wy = (win >> 3) & 7, wx = win & 7;
        int y  = (wy * 8 + (n >> 3) + 4) & 63;
        int xc = (wx * 8 + (n & 7) + 4) & 63;
        srcTok = b * 4096 + y * 64 + xc;
    } else {
        b = r >> 12;
        srcTok = r;
    }
    const float4* sp = (const float4*)(src + (size_t)srcTok * 512);
    float4 vals[4];
    float s = 0.f, ss = 0.f;
#pragma unroll
    for (int i = 0; i < 4; ++i) {
        float4 v = sp[lane + 32 * i];
        vals[i] = v;
        s  += v.x + v.y + v.z + v.w;
        ss += v.x * v.x + v.y * v.y + v.z * v.z + v.w * v.w;
    }
#pragma unroll
    for (int o = 1; o < 32; o <<= 1) {
        s  += __shfl_xor(s, o, 32);
        ss += __shfl_xor(ss, o, 32);
    }
    float mean = s * (1.f / 512.f);
    float var  = ss * (1.f / 512.f) - mean * mean;
    float rstd = rsqrtf(var + 1e-6f);
    const float4* scp = (const float4*)(mod + b * 3072 + scOff);
    const float4* shp = (const float4*)(mod + b * 3072 + shOff);
#pragma unroll
    for (int i = 0; i < 4; ++i) {
        int c4 = lane + 32 * i;
        float4 sc = scp[c4], sh = shp[c4], v = vals[i];
        float o0 = (v.x - mean) * rstd * (1.f + sc.x) + sh.x;
        float o1 = (v.y - mean) * rstd * (1.f + sc.y) + sh.y;
        float o2 = (v.z - mean) * rstd * (1.f + sc.z) + sh.z;
        float o3 = (v.w - mean) * rstd * (1.f + sc.w) + sh.w;
        uint2 pk;
        pk.x = (unsigned)f2bf(o0) | ((unsigned)f2bf(o1) << 16);
        pk.y = (unsigned)f2bf(o2) | ((unsigned)f2bf(o3) << 16);
        *(uint2*)&dst[(size_t)r * 512 + c4 * 4] = pk;
    }
}

// ---------------------------------------------------------------------------
// Tiled bf16 WMMA GEMM: C[M,N] = A[M,K] * Bt[N,K]^T + bias, M = gridDim.y*256
// WG 256 thr = 8 waves (4x2), wave tile 64x32, K-step 32.
// Tile staging via Tensor Data Mover (tensor_load_to_lds + s_wait_tensorcnt)
// when available; sync global->LDS copy (+ global_prefetch) otherwise.
// EPI 0: bf16 store (qkv)         EPI 2: GELU + bf16 store (mlp1)
// EPI 1: window-reverse scatter + residual into f32 out (proj)
// EPI 3: in-place residual f32 out += g * v (mlp2)
// ---------------------------------------------------------------------------
template <int EPI>
__global__ void __launch_bounds__(256) gemm_bf16_kernel(
    const unsigned short* __restrict__ A,   // [M x K] bf16
    const unsigned short* __restrict__ Bt,  // [N x K] bf16 (pre-transposed weights)
    const float* __restrict__ bias,         // [N]
    unsigned short* __restrict__ obf,       // bf16 out (EPI 0/2)
    float* __restrict__ ofp,                // f32 out (EPI 1/3)
    const float* __restrict__ xres,         // residual (EPI 1)
    const float* __restrict__ mod,          // [8 x 3072]
    int N, int K, int goff) {
    __shared__ unsigned short As[256 * 40];   // rows padded to 40 halfwords
    __shared__ unsigned short Bs[64 * 40];

    const int tid  = threadIdx.x;
    const int lane = tid & 31;
    const int wave = tid >> 5;
    const int wm = wave >> 1;   // 0..3
    const int wn = wave & 1;    // 0..1
    const int ml = lane & 15;
    const int hi = lane >> 4;

    const int rowBlock = blockIdx.y * 256;
    const int colBlock = blockIdx.x * 64;

    v8f acc[4][2];
#pragma unroll
    for (int i = 0; i < 4; ++i)
#pragma unroll
        for (int j = 0; j < 2; ++j) acc[i][j] = vzero8();

#if HAS_TDM
    const unsigned asBase = lds_off(&As[0]);
    const unsigned bsBase = lds_off(&Bs[0]);
#else
    const unsigned short* gA = A + (size_t)(rowBlock + tid) * K;
    const unsigned short* gB = Bt + (size_t)(colBlock + (tid >> 2)) * K + (tid & 3) * 8;
#endif

    for (int k0 = 0; k0 < K; k0 += 32) {
#if HAS_TDM
        __syncthreads();  // all LDS reads of previous tile complete
        if (wave == 0) {
            // A tile: 256 rows x 32 elems, row stride K; B tile: 64 rows x 32
            tdm_load_2d(asBase, A + (size_t)rowBlock * K + k0, (unsigned)K, 256u,
                        32u, 256u, (unsigned)K);
            tdm_load_2d(bsBase, Bt + (size_t)colBlock * K + k0, (unsigned)K, 64u,
                        32u, 64u, (unsigned)K);
            __builtin_amdgcn_s_wait_tensorcnt(0);
        }
        __syncthreads();  // tiles visible to all waves
#else
        int4 a0 = *(const int4*)(gA + k0);
        int4 a1 = *(const int4*)(gA + k0 + 8);
        int4 a2 = *(const int4*)(gA + k0 + 16);
        int4 a3 = *(const int4*)(gA + k0 + 24);
        int4 b0 = *(const int4*)(gB + k0);
        if (k0 + 32 < K) {
            __builtin_prefetch((const void*)(gA + k0 + 32), 0, 1);
            __builtin_prefetch((const void*)(gB + k0 + 32), 0, 1);
        }
        __syncthreads();
        *(int4*)&As[tid * 40 + 0]  = a0;
        *(int4*)&As[tid * 40 + 8]  = a1;
        *(int4*)&As[tid * 40 + 16] = a2;
        *(int4*)&As[tid * 40 + 24] = a3;
        *(int4*)&Bs[(tid >> 2) * 40 + (tid & 3) * 8] = b0;
        __syncthreads();
#endif

        Frag af[4], bf[2];
#pragma unroll
        for (int im = 0; im < 4; ++im) {
            const unsigned short* p = &As[(wm * 64 + im * 16 + ml) * 40];
            af[im].i2[0] = *(const int4*)(p + hi * 8);        // K 0-7 / 8-15
            af[im].i2[1] = *(const int4*)(p + 16 + hi * 8);   // K 16-23 / 24-31
        }
#pragma unroll
        for (int in = 0; in < 2; ++in) {
            const unsigned short* p = &Bs[(wn * 32 + in * 16 + ml) * 40];
            bf[in].i2[0] = *(const int4*)(p + hi * 8);
            bf[in].i2[1] = *(const int4*)(p + 16 + hi * 8);
        }
#pragma unroll
        for (int im = 0; im < 4; ++im)
#pragma unroll
            for (int in = 0; in < 2; ++in)
                acc[im][in] = wmma_bf16(af[im].v, bf[in].v, acc[im][in]);
    }

#pragma unroll
    for (int im = 0; im < 4; ++im)
#pragma unroll
        for (int in = 0; in < 2; ++in)
#pragma unroll
            for (int j = 0; j < 8; ++j) {
                int row = rowBlock + wm * 64 + im * 16 + j + hi * 8;
                int col = colBlock + wn * 32 + in * 16 + ml;
                float v = acc[im][in][j] + bias[col];
                if (EPI == 0) {
                    obf[(size_t)row * N + col] = f2bf(v);
                } else if (EPI == 2) {
                    float g = 0.5f * v * (1.0f + erff(v * 0.7071067811865475f));
                    obf[(size_t)row * N + col] = f2bf(g);
                } else if (EPI == 1) {
                    int win = row >> 6, n = row & 63;
                    int b = win >> 6, wy = (win >> 3) & 7, wx = win & 7;
                    int yf = (wy * 8 + (n >> 3) + 4) & 63;
                    int xf = (wx * 8 + (n & 7) + 4) & 63;
                    size_t tgt = ((size_t)b * 4096 + yf * 64 + xf) * 512 + col;
                    float g = mod[b * 3072 + goff + col];
                    ofp[tgt] = xres[tgt] + g * v;
                } else {  // EPI == 3
                    int b = row >> 12;
                    float g = mod[b * 3072 + goff + col];
                    size_t tgt = (size_t)row * 512 + col;
                    ofp[tgt] = ofp[tgt] + g * v;
                }
            }
}

// ---------------------------------------------------------------------------
// K4: windowed attention, one wave per (window, head). N=64, hd=32.
// S = q k^T * scale + relbias; softmax rows; out = P v.  All GEMMs on WMMA.
// ---------------------------------------------------------------------------
__global__ void __launch_bounds__(128) attn_kernel(const unsigned short* __restrict__ qkv,
                                                   const float* __restrict__ rel_bias,
                                                   unsigned short* __restrict__ obf) {
    __shared__ unsigned short S_lds[4][64 * 72];  // P, row stride 72 (pad)
    __shared__ unsigned short vT[4][32 * 72];     // v transposed [d][m]
    __shared__ float bl[4][240];                  // per-head rel bias table (225)

    const int wave = threadIdx.x >> 5, lane = threadIdx.x & 31;
    const int task = blockIdx.x * 4 + wave;
    const int win = task >> 4, head = task & 15;
    const size_t rowBase = (size_t)win * 64;
    const int ml = lane & 15, hi = lane >> 4;

    for (int i = lane; i < 225; i += 32) bl[wave][i] = rel_bias[i * 16 + head];

    // stage V transposed into LDS: vT[d][m] = v[m][d]
#pragma unroll
    for (int t = 0; t < 2; ++t) {
        int m = lane * 2 + t;
        const unsigned* vp = (const unsigned*)(qkv + (rowBase + m) * 1536 + 1024 + head * 32);
#pragma unroll
        for (int i = 0; i < 16; ++i) {
            unsigned u = vp[i];
            vT[wave][(2 * i + 0) * 72 + m] = (unsigned short)(u & 0xffffu);
            vT[wave][(2 * i + 1) * 72 + m] = (unsigned short)(u >> 16);
        }
    }

    // q as A-fragments (4 tiles of 16x32), k as B-fragments (4 tiles of 32x16 = k^T)
    Frag qf[4], kf[4];
#pragma unroll
    for (int tm = 0; tm < 4; ++tm) {
        const unsigned short* p = qkv + (rowBase + tm * 16 + ml) * 1536 + head * 32;
        qf[tm].i2[0] = *(const int4*)(p + hi * 8);
        qf[tm].i2[1] = *(const int4*)(p + 16 + hi * 8);
    }
#pragma unroll
    for (int tn = 0; tn < 4; ++tn) {
        const unsigned short* p =
            qkv + (rowBase + tn * 16 + ml) * 1536 + 512 + head * 32;
        kf[tn].i2[0] = *(const int4*)(p + hi * 8);
        kf[tn].i2[1] = *(const int4*)(p + 16 + hi * 8);
    }

    v8f s[4][4];
#pragma unroll
    for (int tm = 0; tm < 4; ++tm)
#pragma unroll
        for (int tn = 0; tn < 4; ++tn) s[tm][tn] = vzero8();
#pragma unroll
    for (int tm = 0; tm < 4; ++tm)
#pragma unroll
        for (int tn = 0; tn < 4; ++tn)
            s[tm][tn] = wmma_bf16(qf[tm].v, kf[tn].v, s[tm][tn]);

    const float scale = 0.17677669529663687f;  // 1/sqrt(32)
    // softmax along keys; lanes 0-15 own row j, lanes 16-31 own row j+8
#pragma unroll
    for (int tm = 0; tm < 4; ++tm) {
#pragma unroll
        for (int j = 0; j < 8; ++j) {
            int rq = tm * 16 + j + hi * 8;
            int yq = rq >> 3, xq = rq & 7;
            float e[4];
#pragma unroll
            for (int tn = 0; tn < 4; ++tn) {
                int key = tn * 16 + ml;
                int idx = (yq - (key >> 3) + 7) * 15 + (xq - (key & 7) + 7);
                e[tn] = s[tm][tn][j] * scale + bl[wave][idx];
            }
            float mx = fmaxf(fmaxf(e[0], e[1]), fmaxf(e[2], e[3]));
#pragma unroll
            for (int o = 1; o < 16; o <<= 1) mx = fmaxf(mx, __shfl_xor(mx, o, 32));
            float sum = 0.f;
#pragma unroll
            for (int tn = 0; tn < 4; ++tn) { e[tn] = __expf(e[tn] - mx); sum += e[tn]; }
#pragma unroll
            for (int o = 1; o < 16; o <<= 1) sum += __shfl_xor(sum, o, 32);
            float inv = 1.f / sum;
#pragma unroll
            for (int tn = 0; tn < 4; ++tn)
                S_lds[wave][rq * 72 + tn * 16 + ml] = f2bf(e[tn] * inv);
        }
    }

    // out = P (64x64) * v (64x32): A = P tiles from LDS, B = vT tiles from LDS
    v8f o2[4][2];
#pragma unroll
    for (int tm = 0; tm < 4; ++tm)
#pragma unroll
        for (int tn = 0; tn < 2; ++tn) o2[tm][tn] = vzero8();
#pragma unroll
    for (int kc = 0; kc < 2; ++kc) {
        Frag pf[4], vf[2];
#pragma unroll
        for (int tm = 0; tm < 4; ++tm) {
            const unsigned short* p = &S_lds[wave][(tm * 16 + ml) * 72 + kc * 32];
            pf[tm].i2[0] = *(const int4*)(p + hi * 8);
            pf[tm].i2[1] = *(const int4*)(p + 16 + hi * 8);
        }
#pragma unroll
        for (int tn = 0; tn < 2; ++tn) {
            const unsigned short* p = &vT[wave][(tn * 16 + ml) * 72 + kc * 32];
            vf[tn].i2[0] = *(const int4*)(p + hi * 8);
            vf[tn].i2[1] = *(const int4*)(p + 16 + hi * 8);
        }
#pragma unroll
        for (int tm = 0; tm < 4; ++tm)
#pragma unroll
            for (int tn = 0; tn < 2; ++tn)
                o2[tm][tn] = wmma_bf16(pf[tm].v, vf[tn].v, o2[tm][tn]);
    }
#pragma unroll
    for (int tm = 0; tm < 4; ++tm)
#pragma unroll
        for (int tn = 0; tn < 2; ++tn)
#pragma unroll
            for (int j = 0; j < 8; ++j) {
                int rq = tm * 16 + j + hi * 8;
                int col = head * 32 + tn * 16 + ml;
                obf[(rowBase + rq) * 512 + col] = f2bf(o2[tm][tn][j]);
            }
}

// ---------------------------------------------------------------------------
// launch
// ---------------------------------------------------------------------------
extern "C" void kernel_launch(void* const* d_in, const int* in_sizes, int n_in,
                              void* d_out, int out_size, void* d_ws, size_t ws_size,
                              hipStream_t stream) {
    const float* x      = (const float*)d_in[0];
    const float* emb    = (const float*)d_in[1];
    const float* qkv_w  = (const float*)d_in[2];
    const float* qkv_b  = (const float*)d_in[3];
    const float* proj_w = (const float*)d_in[4];
    const float* proj_b = (const float*)d_in[5];
    const float* relb   = (const float*)d_in[6];
    const float* ff_w1  = (const float*)d_in[7];
    const float* ff_b1  = (const float*)d_in[8];
    const float* ff_w2  = (const float*)d_in[9];
    const float* ff_b2  = (const float*)d_in[10];
    const float* ada_w  = (const float*)d_in[11];
    const float* ada_b  = (const float*)d_in[12];
    float* out = (float*)d_out;
    char* ws = (char*)d_ws;

    // workspace layout (bytes)
    constexpr size_t OFF_MOD  = 0;                           //  98,304
    constexpr size_t OFF_WQKV = 98304;                       //  qkv_w^T bf16
    constexpr size_t OFF_WPRJ = OFF_WQKV + 1536u * 512 * 2;  //  proj_w^T
    constexpr size_t OFF_WFF1 = OFF_WPRJ + 512u * 512 * 2;   //  ff_w1^T
    constexpr size_t OFF_WFF2 = OFF_WFF1 + 2048u * 512 * 2;  //  ff_w2^T
    constexpr size_t OFF_HW   = OFF_WFF2 + 512u * 2048 * 2;  //  hw bf16 / h2 bf16 (reused)
    constexpr size_t OFF_BIG  = OFF_HW + 32768u * 512 * 2;   //  qkv bf16 / mlp1 bf16 (reused)
    constexpr size_t OFF_ATT  = OFF_BIG + 32768u * 2048 * 2; //  attn out bf16

    float* mod            = (float*)(ws + OFF_MOD);
    unsigned short* wqkvT = (unsigned short*)(ws + OFF_WQKV);
    unsigned short* wprjT = (unsigned short*)(ws + OFF_WPRJ);
    unsigned short* wff1T = (unsigned short*)(ws + OFF_WFF1);
    unsigned short* wff2T = (unsigned short*)(ws + OFF_WFF2);
    unsigned short* hw    = (unsigned short*)(ws + OFF_HW);   // window-major LN1 out
    unsigned short* h2    = (unsigned short*)(ws + OFF_HW);   // reuse for LN2 out
    unsigned short* qkvb  = (unsigned short*)(ws + OFF_BIG);  // qkv bf16
    unsigned short* mlp1b = (unsigned short*)(ws + OFF_BIG);  // reuse for mlp1 bf16
    unsigned short* attnb = (unsigned short*)(ws + OFF_ATT);

    // K0: weight convert + transpose (bf16)
    wconvT_kernel<<<(1536 * 512 + 255) / 256, 256, 0, stream>>>(qkv_w, wqkvT, 512, 1536);
    wconvT_kernel<<<(512 * 512 + 255) / 256, 256, 0, stream>>>(proj_w, wprjT, 512, 512);
    wconvT_kernel<<<(2048 * 512 + 255) / 256, 256, 0, stream>>>(ff_w1, wff1T, 512, 2048);
    wconvT_kernel<<<(512 * 2048 + 255) / 256, 256, 0, stream>>>(ff_w2, wff2T, 2048, 512);

    // K1: AdaLN modulation
    mod_kernel<<<96, 256, 0, stream>>>(emb, ada_w, ada_b, mod);

    // K2: LN1 + modulate + shifted-window partition -> hw (window-major bf16)
    ln_mod_kernel<<<4096, 256, 0, stream>>>(x, mod, hw, /*sh*/ 0, /*sc*/ 512, 0);

    // K3: qkv = hw @ qkv_w + qkv_b   [32768 x 1536]
    gemm_bf16_kernel<0><<<dim3(24, 128), 256, 0, stream>>>(hw, wqkvT, qkv_b, qkvb,
                                                           nullptr, nullptr, nullptr,
                                                           1536, 512, 0);

    // K4: windowed attention (512 windows x 16 heads, 1 wave each)
    attn_kernel<<<2048, 128, 0, stream>>>(qkvb, relb, attnb);

    // K5: proj + window-reverse + residual -> x1 in d_out
    gemm_bf16_kernel<1><<<dim3(8, 128), 256, 0, stream>>>(attnb, wprjT, proj_b,
                                                          nullptr, out, x, mod,
                                                          512, 512, /*g1*/ 1024);

    // K6: LN2 + modulate -> h2 (token-major bf16), reads x1 from d_out
    ln_mod_kernel<<<4096, 256, 0, stream>>>(out, mod, h2, /*sh2*/ 1536, /*sc2*/ 2048, 1);

    // K7: mlp1 = gelu(h2 @ ff_w1 + ff_b1)   [32768 x 2048]
    gemm_bf16_kernel<2><<<dim3(32, 128), 256, 0, stream>>>(h2, wff1T, ff_b1, mlp1b,
                                                           nullptr, nullptr, nullptr,
                                                           2048, 512, 0);

    // K8: out = x1 + g2 * (mlp1 @ ff_w2 + ff_b2)   (in-place on d_out)
    gemm_bf16_kernel<3><<<dim3(8, 128), 256, 0, stream>>>(mlp1b, wff2T, ff_b2,
                                                          nullptr, out, nullptr, mod,
                                                          512, 2048, /*g2*/ 2560);

    (void)in_sizes; (void)n_in; (void)out_size; (void)ws_size;
}
